// AE_RNN_515396075544
// MI455X (gfx1250) — compile-verified, matrix operand order checked by reading, not run on hardware
//
#include <hip/hip_runtime.h>
#include <hip/hip_bf16.h>
#include <math.h>

// ---------------------------------------------------------------------------
// MI455X (gfx1250) implementation of the VRNN reference.
//
// Roofline: ~350 GFLOP total, ~128 MB of real I/O -> latency-bound by the
// T=2048 sequential GRU recurrence, not bandwidth. So:
//   Phase 1 (parallel over T): phi_u MLP for all (t,b) via WMMA f16->f32.
//   Phase 2 (persistent, sequential): only the GRU chain per step
//            (gi0/gh0 -> GRU0 -> gi1 | gh1 -> GRU1), 4 grid barriers/step.
//            Weight B-fragments are register-resident across all 2048 steps;
//            next step's phi_u tile is prefetched (global_prefetch_b8).
//   Phase 3 (parallel over T): dynn -> x_mean/x_logvar -> phi_x -> menn ->
//            squared-error loss, 7 chained WMMA GEMMs per 16-row tile.
// All matmuls use V_WMMA_F32_16X16X32_F16 (wave32, f32 accumulate).
// ---------------------------------------------------------------------------

typedef __attribute__((ext_vector_type(16))) _Float16 v16h;
typedef __attribute__((ext_vector_type(8)))  float    v8f;

#define Bsz   256
#define Tsz   2048
#define Hsz   128
#define NBLK2 48          // persistent blocks in phase 2
#define NTHR2 (NBLK2*256)

// ------------------------- packed-weight fragment layout -------------------
// One 16(N) x 32(K) B tile = 32 lanes * 16 halves = 512 halves, stored so a
// lane reads its 16 halves contiguously (2x b128 loads).
static constexpr size_t TILE = 512;
static constexpr size_t OFF_PU_W1 = 0;                      //  8 tiles (K=32 ,N=128)
static constexpr size_t OFF_PU_W2 = OFF_PU_W1 +  8*TILE;    // 32 tiles (K=128,N=128)
static constexpr size_t OFF_WIH0  = OFF_PU_W2 + 32*TILE;    // 96 tiles (K=128,N=384)
static constexpr size_t OFF_WHH0  = OFF_WIH0  + 96*TILE;
static constexpr size_t OFF_WIH1  = OFF_WHH0  + 96*TILE;
static constexpr size_t OFF_WHH1  = OFF_WIH1  + 96*TILE;
static constexpr size_t OFF_DYA   = OFF_WHH1  + 96*TILE;    // dy_w1[:, :H]
static constexpr size_t OFF_DYB   = OFF_DYA   + 32*TILE;    // dy_w1[:, H:]
static constexpr size_t OFF_DY2   = OFF_DYB   + 32*TILE;
static constexpr size_t OFF_XM    = OFF_DY2   + 32*TILE;    // 16 tiles (N=64)
static constexpr size_t OFF_XLV   = OFF_XM    + 16*TILE;
static constexpr size_t OFF_PX1   = OFF_XLV   + 16*TILE;
static constexpr size_t OFF_PX2   = OFF_PX1   + 32*TILE;
static constexpr size_t OFF_ME1   = OFF_PX2   + 32*TILE;
static constexpr size_t OFF_ME2   = OFF_ME1   + 32*TILE;    //  8 tiles (N=32)
static constexpr size_t FRAG_HALVES = OFF_ME2 + 8*TILE;

// ------------------------- workspace byte layout ---------------------------
static constexpr size_t WS_CTRL   = 0;                          // barrier cnt/sense
static constexpr size_t WS_FRAGS  = 256;
static constexpr size_t GEMM_B    = (size_t)Bsz*384*4;          // f32 [256][384]
static constexpr size_t WS_GI0    = WS_FRAGS + FRAG_HALVES*2;
static constexpr size_t WS_GH0    = WS_GI0 + GEMM_B;
static constexpr size_t WS_GI1    = WS_GH0 + GEMM_B;
static constexpr size_t WS_GH1    = WS_GI1 + GEMM_B;
static constexpr size_t WS_H0     = WS_GH1 + GEMM_B;            // f16 [256][128]
static constexpr size_t WS_H1     = WS_H0 + (size_t)Bsz*Hsz*2;
static constexpr size_t SEQ_B     = (size_t)Tsz*Bsz*Hsz*2;      // 128 MB each
static constexpr size_t WS_PHIU   = WS_H1 + (size_t)Bsz*Hsz*2;
static constexpr size_t WS_H1HIST = WS_PHIU + SEQ_B;

// ------------------------- WMMA helpers ------------------------------------
__device__ __forceinline__ v8f wmma_f16(v16h a, v16h b, v8f c) {
  // emits v_wmma_f32_16x16x32_f16
  return __builtin_amdgcn_wmma_f32_16x16x32_f16(false, a, false, b,
                                                (short)0, c, false, false);
}

// A fragment: 16 rows x 32 K, row-major f16 with leading dim `ld` (halves).
// 16-bit A layout: lanes 0-15 hold M=0..15 / K-groups {g*8..g*8+7, 16+g*8..}.
__device__ __forceinline__ v16h load_afrag(const _Float16* A, int ld) {
  const int lane = threadIdx.x & 31;
  const int row = lane & 15, g = lane >> 4;
  const _Float16* p = A + (size_t)row * ld + g * 8;
  union { uint4 q[2]; v16h v; } u;
  u.q[0] = *(const uint4*)(p);        // K = g*8   .. g*8+7
  u.q[1] = *(const uint4*)(p + 16);   // K = 16+g*8 .. 16+g*8+7
  return u.v;
}

// B fragment: prepacked, 16 contiguous halves per lane.
__device__ __forceinline__ v16h load_bfrag(const _Float16* tile) {
  const _Float16* p = tile + (size_t)(threadIdx.x & 31) * 16;
  union { uint4 q[2]; v16h v; } u;
  u.q[0] = *(const uint4*)(p);
  u.q[1] = *(const uint4*)(p + 8);
  return u.v;
}

// C/D layout: lane (col = lane&15, g = lane>>4); VGPR r -> row (r + 8g).
__device__ __forceinline__ void store_cfrag_f32(float* out, int ldn, v8f c) {
  const int lane = threadIdx.x & 31;
  const int col = lane & 15, g = lane >> 4;
#pragma unroll
  for (int r = 0; r < 8; ++r)
    out[(size_t)(r + 8 * g) * ldn + col] = c[r];
}

__device__ __forceinline__ float sigmf(float x) {
  return 1.f / (1.f + __expf(-x));
}

// ------------------------- weight packing ----------------------------------
// Packs W[N][ldw] (PyTorch [out,in], used as B = W.T) sub-block starting at
// column k0 into per-lane fragment order. Launch <<<(N/16)*(K/32), 32>>>.
__global__ void pack_weight(const float* __restrict__ W, _Float16* __restrict__ dst,
                            int N, int K, int ldw, int k0) {
  const int tile = blockIdx.x;
  const int ntn = N >> 4;
  const int nt = tile % ntn, kt = tile / ntn;
  const int lane = threadIdx.x & 31;
  const int col = lane & 15, g = lane >> 4;
  const int n = nt * 16 + col;
  _Float16* o = dst + (size_t)tile * TILE + (size_t)lane * 16;
#pragma unroll
  for (int i = 0; i < 8; ++i) {
    const int kb = (i < 4) ? (2 * i + 8 * g) : (16 + 2 * (i - 4) + 8 * g);
    const int k = kt * 32 + kb;
    o[2 * i]     = (_Float16)W[(size_t)n * ldw + k0 + k];
    o[2 * i + 1] = (_Float16)W[(size_t)n * ldw + k0 + k + 1];
  }
}

// init: zero loss + barrier control words, f32->f16 hidden state
__global__ void init_state(const float* __restrict__ h0, char* ws, float* out) {
  int* ctrl = (int*)(ws + WS_CTRL);
  _Float16* h0s = (_Float16*)(ws + WS_H0);
  _Float16* h1s = (_Float16*)(ws + WS_H1);
  if (threadIdx.x == 0) { ctrl[0] = 0; ctrl[1] = 0; out[0] = 0.f; }
  for (int i = threadIdx.x; i < Bsz * Hsz; i += blockDim.x) {
    h0s[i] = (_Float16)h0[i];
    h1s[i] = (_Float16)h0[Bsz * Hsz + i];
  }
}

// ------------------------- Phase 1: phi_u ----------------------------------
// grid <<<Tsz, 256>>>: block = one t, 8 waves cover the 8 N-tiles of H=128.
__global__ void phase1_phiu(const float* __restrict__ u,
                            const float* __restrict__ b1,
                            const float* __restrict__ b2, char* ws) {
  const _Float16* frags = (const _Float16*)(ws + WS_FRAGS);
  _Float16* phiu = (_Float16*)(ws + WS_PHIU);
  __shared__ _Float16 A0[16 * 32];
  __shared__ _Float16 Hid[16 * 128];
  const int t = blockIdx.x;
  const int wid = threadIdx.x >> 5, lane = threadIdx.x & 31;
  const int col = lane & 15, g = lane >> 4;

  // loop-invariant weight fragments (register-resident)
  const v16h wpu1 = load_bfrag(frags + OFF_PU_W1 + (size_t)wid * TILE);
  v16h wpu2[4];
#pragma unroll
  for (int kt = 0; kt < 4; ++kt)
    wpu2[kt] = load_bfrag(frags + OFF_PU_W2 + (size_t)(kt * 8 + wid) * TILE);
  const float bv1 = b1[wid * 16 + col];
  const float bv2 = b2[wid * 16 + col];

  for (int bt = 0; bt < 16; ++bt) {
    // stage u_t tile into LDS as f16 (u is [B][U][T], time fastest)
    for (int e = threadIdx.x; e < 512; e += 256) {
      const int row = e >> 5, k = e & 31;
      A0[e] = (_Float16)u[((size_t)(bt * 16 + row) * 32 + k) * Tsz + t];
    }
    __syncthreads();
    {  // hidden = relu(u @ pu_w1.T + b1), K=32 (one WMMA)
      v8f c = {};
      c = wmma_f16(load_afrag(A0, 32), wpu1, c);
#pragma unroll
      for (int r = 0; r < 8; ++r) {
        float v = c[r] + bv1;
        Hid[(size_t)(r + 8 * g) * 128 + wid * 16 + col] = (_Float16)(v > 0.f ? v : 0.f);
      }
    }
    __syncthreads();
    {  // phi_u = hidden @ pu_w2.T + b2, K=128 (4 WMMAs)
      v8f c = {};
#pragma unroll
      for (int kt = 0; kt < 4; ++kt)
        c = wmma_f16(load_afrag(Hid + kt * 32, 128), wpu2[kt], c);
      _Float16* o = phiu + ((size_t)t * Bsz + bt * 16) * Hsz;
#pragma unroll
      for (int r = 0; r < 8; ++r)
        o[(size_t)(r + 8 * g) * 128 + wid * 16 + col] = (_Float16)(c[r] + bv2);
    }
    __syncthreads();
  }
}

// ------------------------- Phase 2: sequential GRU scan --------------------
__device__ __forceinline__ void grid_barrier(int* cnt, int* sense, int& local) {
  __syncthreads();
  if (threadIdx.x == 0) {
    local ^= 1;
    const int prev = __hip_atomic_fetch_add(cnt, 1, __ATOMIC_ACQ_REL,
                                            __HIP_MEMORY_SCOPE_AGENT);
    if (prev == NBLK2 - 1) {
      __hip_atomic_store(cnt, 0, __ATOMIC_RELAXED, __HIP_MEMORY_SCOPE_AGENT);
      __hip_atomic_store(sense, local, __ATOMIC_RELEASE, __HIP_MEMORY_SCOPE_AGENT);
    } else {
      while (__hip_atomic_load(sense, __ATOMIC_ACQUIRE,
                               __HIP_MEMORY_SCOPE_AGENT) != local)
        __builtin_amdgcn_s_sleep(2);
    }
  }
  __syncthreads();
}

// vectorized GRU elementwise update: 2 channels per iteration
__device__ __forceinline__ void gru_elem(const float* __restrict__ GI,
                                         const float* __restrict__ GH,
                                         _Float16* __restrict__ h,
                                         int tid) {
  unsigned* h32 = (unsigned*)h;
  for (int p = tid; p < Bsz * Hsz / 2; p += NTHR2) {
    const int b = p >> 6;              // 64 channel-pairs per batch row
    const int j = (p & 63) * 2;
    const float* gi = GI + (size_t)b * 384;
    const float* gh = GH + (size_t)b * 384;
    const float2 gir = *(const float2*)(gi + j);
    const float2 giz = *(const float2*)(gi + 128 + j);
    const float2 gin = *(const float2*)(gi + 256 + j);
    const float2 ghr = *(const float2*)(gh + j);
    const float2 ghz = *(const float2*)(gh + 128 + j);
    const float2 ghn = *(const float2*)(gh + 256 + j);
    union { unsigned u; _Float16 f[2]; } ho, hn;
    ho.u = h32[p];
    const float r0 = sigmf(gir.x + ghr.x), r1 = sigmf(gir.y + ghr.y);
    const float z0 = sigmf(giz.x + ghz.x), z1 = sigmf(giz.y + ghz.y);
    const float n0 = tanhf(gin.x + r0 * ghn.x), n1 = tanhf(gin.y + r1 * ghn.y);
    hn.f[0] = (_Float16)((1.f - z0) * n0 + z0 * (float)ho.f[0]);
    hn.f[1] = (_Float16)((1.f - z1) * n1 + z1 * (float)ho.f[1]);
    h32[p] = hn.u;
  }
}

// persistent <<<NBLK2, 256>>>; 384 waves exactly tile the [256 x 384] gates.
__global__ void phase2_scan(char* ws) {
  int* ctrl = (int*)(ws + WS_CTRL);
  const _Float16* frags = (const _Float16*)(ws + WS_FRAGS);
  float* GI0 = (float*)(ws + WS_GI0);
  float* GH0 = (float*)(ws + WS_GH0);
  float* GI1 = (float*)(ws + WS_GI1);
  float* GH1 = (float*)(ws + WS_GH1);
  _Float16* h0s = (_Float16*)(ws + WS_H0);
  _Float16* h1s = (_Float16*)(ws + WS_H1);
  const _Float16* phiu = (const _Float16*)(ws + WS_PHIU);
  _Float16* h1hist = (_Float16*)(ws + WS_H1HIST);

  const int wid = threadIdx.x >> 5;
  const int lane = threadIdx.x & 31;
  const int gw = blockIdx.x * 8 + wid;      // 0..383
  const int bt = gw / 24, nt = gw % 24;     // 16 row tiles x 24 col tiles
  const int tid = blockIdx.x * 256 + threadIdx.x;
  int lsense = 0;

  // Register-cache all weight B-fragments for this wave's (nt) column tile:
  // 16 x v16h = 128 VGPRs. Phase 2 runs ~1 wave/SIMD, so the occupancy cost
  // is irrelevant and the 2048-step loop issues zero weight vmem traffic.
  v16h wih0[4], whh0[4], whh1[4], wih1[4];
#pragma unroll
  for (int kt = 0; kt < 4; ++kt) {
    wih0[kt] = load_bfrag(frags + OFF_WIH0 + (size_t)(kt * 24 + nt) * TILE);
    whh0[kt] = load_bfrag(frags + OFF_WHH0 + (size_t)(kt * 24 + nt) * TILE);
    whh1[kt] = load_bfrag(frags + OFF_WHH1 + (size_t)(kt * 24 + nt) * TILE);
    wih1[kt] = load_bfrag(frags + OFF_WIH1 + (size_t)(kt * 24 + nt) * TILE);
  }

  const _Float16* Ah0 = h0s + (size_t)bt * 16 * Hsz;
  const _Float16* Ah1 = h1s + (size_t)bt * 16 * Hsz;
  float* O_GI0 = GI0 + (size_t)bt * 16 * 384 + nt * 16;
  float* O_GH0 = GH0 + (size_t)bt * 16 * 384 + nt * 16;
  float* O_GI1 = GI1 + (size_t)bt * 16 * 384 + nt * 16;
  float* O_GH1 = GH1 + (size_t)bt * 16 * 384 + nt * 16;

  for (int t = 0; t < Tsz; ++t) {
    const _Float16* Apu = phiu + ((size_t)t * Bsz + bt * 16) * Hsz;
    // ---- S0: gi0 = phi_u@wih0^T, gh0 = h0@whh0^T, gh1 = h1@whh1^T; record h1
    {
      v8f c0 = {}, c1 = {}, c2 = {};
#pragma unroll
      for (int kt = 0; kt < 4; ++kt) {
        c0 = wmma_f16(load_afrag(Apu + kt * 32, 128), wih0[kt], c0);
        c1 = wmma_f16(load_afrag(Ah0 + kt * 32, 128), whh0[kt], c1);
        c2 = wmma_f16(load_afrag(Ah1 + kt * 32, 128), whh1[kt], c2);
      }
      store_cfrag_f32(O_GI0, 384, c0);
      store_cfrag_f32(O_GH0, 384, c1);
      store_cfrag_f32(O_GH1, 384, c2);
      // prefetch next step's phi_u tile (global_prefetch_b8, HBM->L2/L0)
      if (t + 1 < Tsz) {
        const _Float16* nxt = phiu + ((size_t)(t + 1) * Bsz + bt * 16) * Hsz;
        __builtin_prefetch((const void*)(nxt + (size_t)lane * 64), 0, 0);
      }
      // record pre-update h1 for phase 3
      const unsigned* src = (const unsigned*)h1s;
      unsigned* dst = (unsigned*)(h1hist + (size_t)t * Bsz * Hsz);
      for (int i = tid; i < Bsz * Hsz / 2; i += NTHR2) dst[i] = src[i];
    }
    grid_barrier(ctrl, ctrl + 1, lsense);
    // ---- S1: GRU layer 0 elementwise update (in place on h0 f16)
    gru_elem(GI0, GH0, h0s, tid);
    grid_barrier(ctrl, ctrl + 1, lsense);
    // ---- S2: gi1 = h0_new @ wih1^T
    {
      v8f c = {};
#pragma unroll
      for (int kt = 0; kt < 4; ++kt)
        c = wmma_f16(load_afrag(Ah0 + kt * 32, 128), wih1[kt], c);
      store_cfrag_f32(O_GI1, 384, c);
    }
    grid_barrier(ctrl, ctrl + 1, lsense);
    // ---- S3: GRU layer 1 elementwise update
    gru_elem(GI1, GH1, h1s, tid);
    grid_barrier(ctrl, ctrl + 1, lsense);
  }
}

// ------------------------- Phase 3: heads + loss ---------------------------
__device__ __forceinline__ void mlp_tile(const _Float16* A, const _Float16* frags,
                                         size_t off, int ntn, int nt,
                                         const float* bias, bool relu,
                                         _Float16* dst, int colbase) {
  const int lane = threadIdx.x & 31;
  const int col = lane & 15, g = lane >> 4;
  v8f c = {};
#pragma unroll
  for (int kt = 0; kt < 4; ++kt)
    c = wmma_f16(load_afrag(A + kt * 32, 128),
                 load_bfrag(frags + off + (size_t)(kt * ntn + nt) * TILE), c);
  const float bv = bias[nt * 16 + col];
#pragma unroll
  for (int r = 0; r < 8; ++r) {
    float v = c[r] + bv;
    if (relu) v = v > 0.f ? v : 0.f;
    dst[(size_t)(r + 8 * g) * 128 + colbase + col] = (_Float16)v;
  }
}

// grid <<<Tsz, 256>>>: block = one t; chain of 7 GEMMs per 16-row tile.
__global__ void phase3_heads(char* ws, const float* __restrict__ y,
                             const float* db1, const float* db2,
                             const float* xmb, const float* xlvb,
                             const float* pb1, const float* pb2,
                             const float* mb1, const float* mb2,
                             float* __restrict__ out) {
  const _Float16* frags = (const _Float16*)(ws + WS_FRAGS);
  const _Float16* phiu = (const _Float16*)(ws + WS_PHIU);
  const _Float16* h1hist = (const _Float16*)(ws + WS_H1HIST);
  __shared__ _Float16 buf0[16 * 128];
  __shared__ _Float16 buf1[16 * 128];
  __shared__ float red[256];
  const int t = blockIdx.x;
  const int wid = threadIdx.x >> 5, lane = threadIdx.x & 31;
  const int col = lane & 15, g = lane >> 4;
  float acc = 0.f;

  for (int bt = 0; bt < 16; ++bt) {
    const _Float16* Apu = phiu + ((size_t)t * Bsz + bt * 16) * Hsz;
    const _Float16* Ah1 = h1hist + ((size_t)t * Bsz + bt * 16) * Hsz;
    {  // dynn hidden = relu([phi_u, h1]@dy_w1^T + b1)  -> buf0
      v8f c = {};
#pragma unroll
      for (int kt = 0; kt < 4; ++kt) {
        c = wmma_f16(load_afrag(Apu + kt * 32, 128),
                     load_bfrag(frags + OFF_DYA + (size_t)(kt * 8 + wid) * TILE), c);
        c = wmma_f16(load_afrag(Ah1 + kt * 32, 128),
                     load_bfrag(frags + OFF_DYB + (size_t)(kt * 8 + wid) * TILE), c);
      }
      const float bv = db1[wid * 16 + col];
#pragma unroll
      for (int r = 0; r < 8; ++r) {
        float v = c[r] + bv;
        buf0[(size_t)(r + 8 * g) * 128 + wid * 16 + col] = (_Float16)(v > 0.f ? v : 0.f);
      }
    }
    __syncthreads();
    mlp_tile(buf0, frags, OFF_DY2, 8, wid, db2, false, buf1, wid * 16);  // dphi
    __syncthreads();
    {  // heads: waves 0-3 -> x_mean (cols 0..63); waves 4-7 -> x_logvar (64..127)
      const int nt = wid & 3;
      const size_t base = (wid < 4) ? OFF_XM : OFF_XLV;
      const float* bb = (wid < 4) ? xmb : xlvb;
      const int cb = ((wid < 4) ? 0 : 64) + nt * 16;
      v8f c = {};
#pragma unroll
      for (int kt = 0; kt < 4; ++kt)
        c = wmma_f16(load_afrag(buf1 + kt * 32, 128),
                     load_bfrag(frags + base + (size_t)(kt * 4 + nt) * TILE), c);
      const float bv = bb[nt * 16 + col];
#pragma unroll
      for (int r = 0; r < 8; ++r)
        buf0[(size_t)(r + 8 * g) * 128 + cb + col] = (_Float16)(c[r] + bv);
    }
    __syncthreads();
    mlp_tile(buf0, frags, OFF_PX1, 8, wid, pb1, true, buf1, wid * 16);   // phi_x hid
    __syncthreads();
    mlp_tile(buf1, frags, OFF_PX2, 8, wid, pb2, false, buf0, wid * 16);  // phi_x
    __syncthreads();
    mlp_tile(buf0, frags, OFF_ME1, 8, wid, mb1, true, buf1, wid * 16);   // menn hid
    __syncthreads();
    if (wid < 2) {  // y_hat (Y=32 -> 2 col tiles) + squared-error accumulate
      v8f c = {};
#pragma unroll
      for (int kt = 0; kt < 4; ++kt)
        c = wmma_f16(load_afrag(buf1 + kt * 32, 128),
                     load_bfrag(frags + OFF_ME2 + (size_t)(kt * 2 + wid) * TILE), c);
      const float bv = mb2[wid * 16 + col];
#pragma unroll
      for (int r = 0; r < 8; ++r) {
        const int b = bt * 16 + r + 8 * g;
        const int n = wid * 16 + col;
        const float d = c[r] + bv - y[((size_t)b * 32 + n) * Tsz + t];
        acc += d * d;
      }
    }
    __syncthreads();
  }
  red[threadIdx.x] = acc;
  __syncthreads();
  for (int s = 128; s > 0; s >>= 1) {
    if (threadIdx.x < s) red[threadIdx.x] += red[threadIdx.x + s];
    __syncthreads();
  }
  if (threadIdx.x == 0) atomicAdd(out, red[0]);
}

// ---------------------------------------------------------------------------
extern "C" void kernel_launch(void* const* d_in, const int* in_sizes, int n_in,
                              void* d_out, int out_size, void* d_ws, size_t ws_size,
                              hipStream_t stream) {
  const float* u      = (const float*)d_in[0];
  const float* y      = (const float*)d_in[1];
  const float* h0     = (const float*)d_in[2];
  const float* pu_w1  = (const float*)d_in[3];
  const float* pu_b1  = (const float*)d_in[4];
  const float* pu_w2  = (const float*)d_in[5];
  const float* pu_b2  = (const float*)d_in[6];
  const float* dy_w1  = (const float*)d_in[7];
  const float* dy_b1  = (const float*)d_in[8];
  const float* dy_w2  = (const float*)d_in[9];
  const float* dy_b2  = (const float*)d_in[10];
  const float* xm_w   = (const float*)d_in[11];
  const float* xm_b   = (const float*)d_in[12];
  const float* xlv_w  = (const float*)d_in[13];
  const float* xlv_b  = (const float*)d_in[14];
  const float* px_w1  = (const float*)d_in[15];
  const float* px_b1  = (const float*)d_in[16];
  const float* px_w2  = (const float*)d_in[17];
  const float* px_b2  = (const float*)d_in[18];
  const float* me_w1  = (const float*)d_in[19];
  const float* me_b1  = (const float*)d_in[20];
  const float* me_w2  = (const float*)d_in[21];
  const float* me_b2  = (const float*)d_in[22];
  const float* gru_wih = (const float*)d_in[23];  // [2][384][128]
  const float* gru_whh = (const float*)d_in[24];

  char* ws = (char*)d_ws;
  _Float16* frags = (_Float16*)(ws + WS_FRAGS);
  float* out = (float*)d_out;

  init_state<<<1, 256, 0, stream>>>(h0, ws, out);

  pack_weight<<< 8, 32, 0, stream>>>(pu_w1, frags + OFF_PU_W1, 128,  32,  32, 0);
  pack_weight<<<32, 32, 0, stream>>>(pu_w2, frags + OFF_PU_W2, 128, 128, 128, 0);
  pack_weight<<<96, 32, 0, stream>>>(gru_wih,          frags + OFF_WIH0, 384, 128, 128, 0);
  pack_weight<<<96, 32, 0, stream>>>(gru_whh,          frags + OFF_WHH0, 384, 128, 128, 0);
  pack_weight<<<96, 32, 0, stream>>>(gru_wih + 49152,  frags + OFF_WIH1, 384, 128, 128, 0);
  pack_weight<<<96, 32, 0, stream>>>(gru_whh + 49152,  frags + OFF_WHH1, 384, 128, 128, 0);
  pack_weight<<<32, 32, 0, stream>>>(dy_w1, frags + OFF_DYA, 128, 128, 256, 0);
  pack_weight<<<32, 32, 0, stream>>>(dy_w1, frags + OFF_DYB, 128, 128, 256, 128);
  pack_weight<<<32, 32, 0, stream>>>(dy_w2, frags + OFF_DY2, 128, 128, 128, 0);
  pack_weight<<<16, 32, 0, stream>>>(xm_w,  frags + OFF_XM,   64, 128, 128, 0);
  pack_weight<<<16, 32, 0, stream>>>(xlv_w, frags + OFF_XLV,  64, 128, 128, 0);
  pack_weight<<<32, 32, 0, stream>>>(px_w1, frags + OFF_PX1, 128, 128, 128, 0);
  pack_weight<<<32, 32, 0, stream>>>(px_w2, frags + OFF_PX2, 128, 128, 128, 0);
  pack_weight<<<32, 32, 0, stream>>>(me_w1, frags + OFF_ME1, 128, 128, 128, 0);
  pack_weight<<< 8, 32, 0, stream>>>(me_w2, frags + OFF_ME2,  32, 128, 128, 0);

  phase1_phiu<<<Tsz, 256, 0, stream>>>(u, pu_b1, pu_b2, ws);
  phase2_scan<<<NBLK2, 256, 0, stream>>>(ws);
  phase3_heads<<<Tsz, 256, 0, stream>>>(ws, y, dy_b1, dy_b2, xm_b, xlv_b,
                                        px_b1, px_b2, me_b1, me_b2, out);
}
